// BaselineAttention_36172214567310
// MI455X (gfx1250) — matrix operand order repaired and checked by reading, not compiled, not run on hardware
//
#include <hip/hip_runtime.h>
#include <hip/hip_bf16.h>
#include <stdint.h>

// Problem dims (fixed by the reference)
#define HIDDEN  1024
#define NHEADS  16
#define HDIM    64
#define BB      2
#define SSEQ    2048
#define BSROWS  (BB * SSEQ)     // 4096
#define F3      3072            // 3 * NHEADS * HDIM

typedef __attribute__((ext_vector_type(16))) __bf16 v16bf;
typedef __attribute__((ext_vector_type(8)))  __bf16 v8bf;
typedef __attribute__((ext_vector_type(8)))  float  v8f;
typedef __attribute__((ext_vector_type(4)))  unsigned int v4u;
typedef __attribute__((ext_vector_type(8)))  int    v8i;
typedef __attribute__((ext_vector_type(4)))  int    v4i;

#if defined(__has_builtin)
#if __has_builtin(__builtin_amdgcn_tensor_load_to_lds)
#define HAVE_TDM 1
#else
#define HAVE_TDM 0
#endif
#else
#define HAVE_TDM 0
#endif

// ---------- scalar conversion helpers (RNE f32<->bf16 via bit ops) ----------
static __device__ __forceinline__ __bf16 f2bf(float f) {
    uint32_t u = __builtin_bit_cast(uint32_t, f);
    uint32_t r = u + 0x7FFFu + ((u >> 16) & 1u);
    return __builtin_bit_cast(__bf16, (uint16_t)(r >> 16));
}
static __device__ __forceinline__ float bf2f(__bf16 h) {
    uint32_t u = ((uint32_t)__builtin_bit_cast(uint16_t, h)) << 16;
    return __builtin_bit_cast(float, u);
}

// ---------- WMMA fragment loaders (CDNA5 wave32 operand layouts) ----------
// A-matrix 16x32 bf16: lane holds row M=(lane&15); K pattern per lane is
// kbase+{0..7} then kbase+16+{0..7} (kbase=(lane>>4)*8): elements p[0..7],p[16..23]
static __device__ __forceinline__ v16bf frag_gap16(const __bf16* p) {
    v8bf lo = *(const v8bf*)p;
    v8bf hi = *(const v8bf*)(p + 16);
    return __builtin_shufflevector(lo, hi, 0,1,2,3,4,5,6,7,8,9,10,11,12,13,14,15);
}
// contiguous-k loader (unused here): lane holds column N=(lane&15); K=(lane>>4)*16 + e, e=0..15
// contiguous -> 16 contiguous bf16 from a K-major (column) buffer
static __device__ __forceinline__ v16bf frag_ctg16(const __bf16* p) {
    v8bf lo = *(const v8bf*)p;
    v8bf hi = *(const v8bf*)(p + 8);
    return __builtin_shufflevector(lo, hi, 0,1,2,3,4,5,6,7,8,9,10,11,12,13,14,15);
}
static __device__ __forceinline__ v8f wmma_bf16(v16bf a, v16bf b, v8f c) {
    return __builtin_amdgcn_wmma_f32_16x16x32_bf16(false, a, false, b, (short)0, c, false, false);
}
static __device__ __forceinline__ v8f zero8() {
    v8f z = {0.f,0.f,0.f,0.f,0.f,0.f,0.f,0.f};
    return z;
}

// ---------- Tensor Data Mover: 2-D tile, global -> LDS ----------
static __device__ __forceinline__ void wait_tensorcnt0() {
#if defined(__has_builtin) && __has_builtin(__builtin_amdgcn_s_wait_tensorcnt)
    __builtin_amdgcn_s_wait_tensorcnt(0);
#else
    asm volatile("s_wait_tensorcnt 0x0" ::: "memory");
#endif
}

#if HAVE_TDM
// Build D# per CDNA5 ISA §8 and issue TENSOR_LOAD_TO_LDS.
// data_size is hardcoded to 2 bytes (bf16). Tile is width x height elements,
// global rows strided by pitch elements; LDS destination is packed row-major.
static __device__ __forceinline__ void tdm_load_2d_bf16(uint32_t lds_byte_addr,
                                                        const void* gptr,
                                                        uint32_t width,
                                                        uint32_t height,
                                                        uint32_t pitch) {
    uint64_t ga = (uint64_t)(uintptr_t)gptr;
    v4u g0;
    g0[0] = 1u;                                      // count=1, is_restore=0, no gather
    g0[1] = lds_byte_addr;                           // lds_addr  [63:32]
    g0[2] = (uint32_t)(ga & 0xFFFFFFFFu);            // global_addr[31:0]
    g0[3] = (uint32_t)((ga >> 32) & 0x01FFFFFFu)     // global_addr[56:32]
          | (2u << 30);                              // type=2 ("image")
    v8i g1;
    g1[0] = (int)(1u << 16);                         // wg_mask=0, data_size=1 (2B)
    g1[1] = (int)((width & 0xFFFFu) << 16);          // tensor_dim0[15:0] @bit48
    g1[2] = (int)(((width >> 16) & 0xFFFFu)          // tensor_dim0[31:16]
          |       ((height & 0xFFFFu) << 16));       // tensor_dim1[15:0]
    g1[3] = (int)(((height >> 16) & 0xFFFFu)         // tensor_dim1[31:16]
          |       ((width & 0xFFFFu) << 16));        // tile_dim0 @bit112
    g1[4] = (int)(height & 0xFFFFu);                 // tile_dim1; tile_dim2=0 (2-D)
    g1[5] = (int)pitch;                              // tensor_dim0_stride[31:0]
    g1[6] = 0;                                       // stride[47:32], dim1_stride[15:0]
    g1[7] = 0;                                       // dim1_stride[47:16]
    v4i z4 = {0, 0, 0, 0};
#if __clang_major__ >= 23
    v8i z8 = {0, 0, 0, 0, 0, 0, 0, 0};
    __builtin_amdgcn_tensor_load_to_lds(g0, g1, z4, z4, z8, 0);
#else
    __builtin_amdgcn_tensor_load_to_lds(g0, g1, z4, z4, 0);
#endif
}
#endif

static __device__ __forceinline__ uint32_t lds_off(const void* p) {
    // LDS aperture: hardware truncates generic address to addr[31:0]
    return (uint32_t)(uintptr_t)p;
}

// ---------- elementwise kernels ----------
__global__ void f32_to_bf16_kernel(const float* __restrict__ src,
                                   __bf16* __restrict__ dst, int n) {
    int i = blockIdx.x * blockDim.x + threadIdx.x;
    int stride = gridDim.x * blockDim.x;
    for (; i < n; i += stride) dst[i] = f2bf(src[i]);
}

__global__ void zero_f32_kernel(float* __restrict__ p, int n) {
    int i = blockIdx.x * blockDim.x + threadIdx.x;
    int stride = gridDim.x * blockDim.x;
    for (; i < n; i += stride) p[i] = 0.0f;
}

// vscaled[bs][c] = colsum[b][h][s] * V(bf16)   (c = h*64 + d)
__global__ void scale_v_kernel(const __bf16* __restrict__ qkvb,
                               const float* __restrict__ colsum,
                               __bf16* __restrict__ vsc) {
    int i = blockIdx.x * blockDim.x + threadIdx.x;   // [0, BSROWS*HIDDEN)
    int c  = i & (HIDDEN - 1);
    int bs = i >> 10;
    int b  = bs >> 11;
    int s  = bs & (SSEQ - 1);
    int h  = c >> 6;
    float v  = bf2f(qkvb[(size_t)bs * F3 + 2 * HIDDEN + c]);
    float cs = colsum[(((size_t)b * NHEADS + h) << 11) + s];
    vsc[i] = f2bf(v * cs);
}

// ---------- tiled bf16 WMMA GEMM: C[M,N] = A[M,K] * B[K,N] ----------
// 128x128 block tile, BK=32, double-buffered LDS. 256 threads = 8 waves;
// wave w covers rows (w&1)*64..+63 (4 m-frags) and cols (w>>1)*32..+31
// (2 n-frags). A tile staged by the Tensor Data Mover (wave 0, async,
// overlapped with WMMA on the other buffer); B tile staged manually
// transposed (TDM cannot transpose).
template <bool OUT_BF16>
__global__ void gemm_bf16_wmma(const __bf16* __restrict__ A,
                               const __bf16* __restrict__ Bm,
                               void* __restrict__ Cout,
                               int M, int N, int K) {
    __shared__ __bf16 As[2][128 * 32];   // row-major [m][k]
    __shared__ __bf16 BsT[2][128 * 32];  // transposed [n][k]

    const int tid  = threadIdx.x;
    const int lane = tid & 31;
    const int wave = tid >> 5;
    const int m0 = blockIdx.y * 128;
    const int n0 = blockIdx.x * 128;
    const int mq = (wave & 1) * 64;
    const int nb = (wave >> 1) * 32;

    auto stageA = [&](int k0, int buf) {
#if HAVE_TDM
        if (wave == 0)
            tdm_load_2d_bf16(lds_off(&As[buf][0]), &A[(size_t)m0 * K + k0],
                             32u, 128u, (uint32_t)K);
#else
#pragma unroll
        for (int j = 0; j < 2; ++j) {
            int c = tid + j * 256;
            int r = c >> 2, cg = c & 3;
            *(v8bf*)&As[buf][r * 32 + cg * 8] =
                *(const v8bf*)&A[(size_t)(m0 + r) * K + k0 + cg * 8];
        }
#endif
    };
    auto stageB = [&](int k0, int buf) {
#pragma unroll
        for (int j = 0; j < 2; ++j) {
            int c = tid + j * 256;
            int kr = c >> 4, ng = c & 15;
            v8bf v = *(const v8bf*)&Bm[(size_t)(k0 + kr) * N + n0 + ng * 8];
#pragma unroll
            for (int e = 0; e < 8; ++e) BsT[buf][(ng * 8 + e) * 32 + kr] = v[e];
        }
    };
    auto fence = [&]() {
#if HAVE_TDM
        if (wave == 0) wait_tensorcnt0();   // publish in-flight TDM tile
#endif
        __syncthreads();
    };

    v8f acc[4][2];
#pragma unroll
    for (int mf = 0; mf < 4; ++mf)
#pragma unroll
        for (int nf = 0; nf < 2; ++nf) acc[mf][nf] = zero8();

    // pipeline prologue: tile 0 into buffer 0
    stageA(0, 0);
    stageB(0, 0);
    fence();

    const int KT = K / 32;
    for (int t = 0; t < KT; ++t) {
        const int cur = t & 1;
        if (t + 1 < KT) {
            // async-stage next tile into the other buffer while computing
            stageA((t + 1) * 32, cur ^ 1);
            stageB((t + 1) * 32, cur ^ 1);
            if (t + 2 < KT)   // keep a cacheline prefetch ahead of staging
                __builtin_prefetch(&Bm[(size_t)((t + 2) * 32 + (tid >> 4)) * N
                                       + n0 + (tid & 15) * 8], 0, 0);
        }
        v16bf af[4];
#pragma unroll
        for (int mf = 0; mf < 4; ++mf)
            af[mf] = frag_gap16(&As[cur][(mq + mf * 16 + (lane & 15)) * 32 + (lane >> 4) * 8]);
        v16bf bfr[2];
#pragma unroll
        for (int nf = 0; nf < 2; ++nf)
            bfr[nf] = frag_gap16(&BsT[cur][(nb + nf * 16 + (lane & 15)) * 32 + (lane >> 4) * 8]);
#pragma unroll
        for (int mf = 0; mf < 4; ++mf)
#pragma unroll
            for (int nf = 0; nf < 2; ++nf)
                acc[mf][nf] = wmma_bf16(af[mf], bfr[nf], acc[mf][nf]);
        fence();
    }

    // epilogue: C/D layout — lane holds col N=(lane&15), row = vgpr + (lane>>4)*8
#pragma unroll
    for (int mf = 0; mf < 4; ++mf)
#pragma unroll
        for (int nf = 0; nf < 2; ++nf) {
            int col = n0 + nb + nf * 16 + (lane & 15);
#pragma unroll
            for (int i = 0; i < 8; ++i) {
                int row = m0 + mq + mf * 16 + (lane >> 4) * 8 + i;
                float v = acc[mf][nf][i];
                if (OUT_BF16)
                    ((__bf16*)Cout)[(size_t)row * N + col] = f2bf(v);
                else
                    ((float*)Cout)[(size_t)row * N + col] = v;
            }
        }
}

// ---------- attention column-sum kernel ----------
// One block per (b, h, 64-row q-tile). Computes colsum[b,h,k] += sum_q P[q,k]
// with P = softmax_k(Q K^T / 8). 3 passes over K tiles: row-max, row-denominator,
// probabilities + column sums (colsum is k-indexed, so flash rescale doesn't apply).
// K tiles are double-buffered and DMA'd by the Tensor Data Mover, overlapping
// the WMMA compute and the softmax reductions.
__global__ void attn_colsum_kernel(const __bf16* __restrict__ qkvb,
                                   float* __restrict__ colsum) {
    const int qt = blockIdx.x;   // 0..31
    const int h  = blockIdx.y;   // 0..15
    const int b  = blockIdx.z;   // 0..1
    const int tid  = threadIdx.x;
    const int lane = tid & 31;
    const int wave = tid >> 5;
    const int q0 = qt * 64;
    const size_t rowBase = (size_t)b * SSEQ;
    const int NT = SSEQ / 128;   // 16 K tiles per pass

    __shared__ __bf16 Qs[64 * 64];       // [q][d]
    __shared__ __bf16 Ks[2][128 * 64];   // [k][d] double-buffered
    __shared__ float  mLds[64];
    __shared__ float  zLds[64];

    const __bf16* kGlobal = qkvb + HIDDEN + h * HDIM;   // K section of qkv

    auto stage_k = [&](int kt, int buf) {
#if HAVE_TDM
        if (wave == 0)
            tdm_load_2d_bf16(lds_off(&Ks[buf][0]),
                             &kGlobal[(rowBase + (size_t)kt * 128) * F3],
                             64u, 128u, (uint32_t)F3);
#else
#pragma unroll
        for (int j = 0; j < 4; ++j) {
            int c = tid + j * 256;
            int r = c >> 3, cg = c & 7;
            *(v8bf*)&Ks[buf][r * 64 + cg * 8] =
                *(const v8bf*)&kGlobal[(rowBase + (size_t)kt * 128 + r) * F3 + cg * 8];
        }
#endif
    };
    auto fence = [&]() {
#if HAVE_TDM
        if (wave == 0) wait_tensorcnt0();
#endif
        __syncthreads();
    };

    // stage Q tile (64x64 bf16) + first K tile together
#if HAVE_TDM
    if (wave == 0)
        tdm_load_2d_bf16(lds_off(Qs), &qkvb[(rowBase + q0) * F3 + h * HDIM],
                         64u, 64u, (uint32_t)F3);
#else
#pragma unroll
    for (int j = 0; j < 2; ++j) {
        int c = tid + j * 256;
        int r = c >> 3, cg = c & 7;
        *(v8bf*)&Qs[r * 64 + cg * 8] =
            *(const v8bf*)&qkvb[(rowBase + q0 + r) * F3 + h * HDIM + cg * 8];
    }
#endif
    stage_k(0, 0);
    fence();

    const int qb   = (wave & 3) * 16;     // wave's 16 q rows
    const int kwb  = (wave >> 2) * 64;    // wave's k-column base within 128-tile
    const int rsel = (lane >> 4) * 8;     // row sub-base within frag

    // Q stays resident as A-fragments for the whole kernel
    v16bf aq[2];
    aq[0] = frag_gap16(&Qs[(qb + (lane & 15)) * 64 +  0 + (lane >> 4) * 8]);
    aq[1] = frag_gap16(&Qs[(qb + (lane & 15)) * 64 + 32 + (lane >> 4) * 8]);

    auto compute_acc = [&](v8f acc[4], int buf) {
#pragma unroll
        for (int f = 0; f < 4; ++f) {
            acc[f] = zero8();
#pragma unroll
            for (int ks = 0; ks < 2; ++ks) {
                v16bf bk = frag_gap16(
                    &Ks[buf][(kwb + f * 16 + (lane & 15)) * 64 + ks * 32 + (lane >> 4) * 8]);
                acc[f] = wmma_bf16(aq[ks], bk, acc[f]);
            }
        }
    };

    // ---- pass 1: row max ----
    float rmax[8];
#pragma unroll
    for (int i = 0; i < 8; ++i) rmax[i] = -1e30f;
    for (int kt = 0; kt < NT; ++kt) {
        const int cur = kt & 1;
        if (kt + 1 < NT) stage_k(kt + 1, cur ^ 1);
        v8f acc[4];
        compute_acc(acc, cur);
#pragma unroll
        for (int f = 0; f < 4; ++f)
#pragma unroll
            for (int i = 0; i < 8; ++i)
                rmax[i] = fmaxf(rmax[i], acc[f][i] * 0.125f);
        fence();
    }
    stage_k(0, 0);   // pass-2 prologue DMA overlaps the reduction below
#pragma unroll
    for (int mask = 1; mask <= 8; mask <<= 1)
#pragma unroll
        for (int i = 0; i < 8; ++i)
            rmax[i] = fmaxf(rmax[i], __shfl_xor(rmax[i], mask, 32));
    if (wave < 4 && (lane & 15) == 0)
#pragma unroll
        for (int i = 0; i < 8; ++i) mLds[qb + rsel + i] = rmax[i];
    __syncthreads();
    if (wave >= 4 && (lane & 15) == 0)
#pragma unroll
        for (int i = 0; i < 8; ++i)
            mLds[qb + rsel + i] = fmaxf(mLds[qb + rsel + i], rmax[i]);
    __syncthreads();
    float rm[8];
#pragma unroll
    for (int i = 0; i < 8; ++i) rm[i] = mLds[qb + rsel + i];
    fence();

    // ---- pass 2: row denominator ----
    float zsum[8];
#pragma unroll
    for (int i = 0; i < 8; ++i) zsum[i] = 0.0f;
    for (int kt = 0; kt < NT; ++kt) {
        const int cur = kt & 1;
        if (kt + 1 < NT) stage_k(kt + 1, cur ^ 1);
        v8f acc[4];
        compute_acc(acc, cur);
#pragma unroll
        for (int f = 0; f < 4; ++f)
#pragma unroll
            for (int i = 0; i < 8; ++i)
                zsum[i] += __expf(acc[f][i] * 0.125f - rm[i]);
        fence();
    }
    stage_k(0, 0);   // pass-3 prologue DMA overlaps the reduction below
#pragma unroll
    for (int mask = 1; mask <= 8; mask <<= 1)
#pragma unroll
        for (int i = 0; i < 8; ++i)
            zsum[i] += __shfl_xor(zsum[i], mask, 32);
    if (wave < 4 && (lane & 15) == 0)
#pragma unroll
        for (int i = 0; i < 8; ++i) zLds[qb + rsel + i] = zsum[i];
    __syncthreads();
    if (wave >= 4 && (lane & 15) == 0)
#pragma unroll
        for (int i = 0; i < 8; ++i) zLds[qb + rsel + i] += zsum[i];
    __syncthreads();
    float rzi[8];
#pragma unroll
    for (int i = 0; i < 8; ++i) rzi[i] = 1.0f / zLds[qb + rsel + i];
    fence();

    // ---- pass 3: probabilities + column sums ----
    float* csBase = colsum + (((size_t)b * NHEADS + h) << 11);
    for (int kt = 0; kt < NT; ++kt) {
        const int cur = kt & 1;
        if (kt + 1 < NT) stage_k(kt + 1, cur ^ 1);
        v8f acc[4];
        compute_acc(acc, cur);
#pragma unroll
        for (int f = 0; f < 4; ++f) {
            float part = 0.0f;
#pragma unroll
            for (int i = 0; i < 8; ++i)
                part += __expf(acc[f][i] * 0.125f - rm[i]) * rzi[i];
            part += __shfl_xor(part, 16, 32);   // combine rows 0-7 and 8-15 halves
            if (lane < 16)
                atomicAdd(&csBase[kt * 128 + kwb + f * 16 + lane], part);
        }
        fence();
    }
}

// ---------- host launcher ----------
extern "C" void kernel_launch(void* const* d_in, const int* in_sizes, int n_in,
                              void* d_out, int out_size, void* d_ws, size_t ws_size,
                              hipStream_t stream) {
    (void)in_sizes; (void)n_in; (void)out_size; (void)ws_size;
    const float* x    = (const float*)d_in[0];
    const float* wqkv = (const float*)d_in[1];
    const float* wo   = (const float*)d_in[2];

    char* ws = (char*)d_ws;
    const size_t MB = 1024u * 1024u;
    __bf16* xb     = (__bf16*)(ws + 0);        //  8 MiB: [4096,1024] bf16
    __bf16* wqkvb  = (__bf16*)(ws + 8 * MB);   //  6 MiB: [1024,3072] bf16
    __bf16* wob    = (__bf16*)(ws + 14 * MB);  //  2 MiB: [1024,1024] bf16
    __bf16* qkvb   = (__bf16*)(ws + 16 * MB);  // 24 MiB: [4096,3072] bf16
    float*  colsum = (float*) (ws + 40 * MB);  // 256 KiB: [2,16,2048] f32
    __bf16* vsc    = (__bf16*)(ws + 41 * MB);  //  8 MiB: [4096,1024] bf16

    // 1) downconvert inputs to bf16
    f32_to_bf16_kernel<<<512, 256, 0, stream>>>(x, xb, BSROWS * HIDDEN);
    f32_to_bf16_kernel<<<512, 256, 0, stream>>>(wqkv, wqkvb, HIDDEN * F3);
    f32_to_bf16_kernel<<<512, 256, 0, stream>>>(wo, wob, HIDDEN * HIDDEN);

    // 2) QKV projection: qkv = x @ w_qkv   (bf16 out, f32 accum)
    dim3 g1(F3 / 128, BSROWS / 128);
    gemm_bf16_wmma<true><<<g1, 256, 0, stream>>>(xb, wqkvb, (void*)qkvb,
                                                 BSROWS, F3, HIDDEN);

    // 3) colsum[b,h,k] = sum_q softmax(QK^T/8)[q,k]
    zero_f32_kernel<<<64, 256, 0, stream>>>(colsum, BB * NHEADS * SSEQ);
    dim3 ga(SSEQ / 64, NHEADS, BB);
    attn_colsum_kernel<<<ga, 256, 0, stream>>>(qkvb, colsum);

    // 4) values = colsum ⊙ V (the buggy einsum collapses P·V to this)
    scale_v_kernel<<<(BSROWS * HIDDEN) / 256, 256, 0, stream>>>(qkvb, colsum, vsc);

    // 5) output projection: out = values @ w_o  (f32 out)
    dim3 g2(HIDDEN / 128, BSROWS / 128);
    gemm_bf16_wmma<false><<<g2, 256, 0, stream>>>(vsc, wob, d_out,
                                                  BSROWS, HIDDEN, HIDDEN);
}
